// Solver_46394236731449
// MI455X (gfx1250) — compile-verified, matrix operand order checked by reading, not run on hardware
//
#include <hip/hip_runtime.h>
#include <hip/hip_bf16.h>
#include <math.h>

typedef __attribute__((ext_vector_type(16))) _Float16 v16h;
typedef __attribute__((ext_vector_type(8)))  float    v8f;

namespace {
constexpr int   kPaths = 65536;
constexpr int   kHid   = 64;
constexpr int   kSteps = 32;
constexpr float kDt    = 1.0f / 32.0f;
constexpr float kMuC   = 0.05f;
constexpr float kSigC  = 0.2f;
constexpr float kTwoLog2E = 2.8853900817779268f;          // 2*log2(e)
constexpr int   kWavesPerBlock = 8;                       // 256 threads, wave32
constexpr int   kPathsPerWave  = 16;                      // one WMMA N-tile
constexpr int   kPathsPerBlock = kWavesPerBlock * kPathsPerWave; // 128
constexpr int   kBlocks        = kPaths / kPathsPerBlock;        // 512
}

#if __has_builtin(__builtin_amdgcn_exp2f)
#define EXP2F(x) __builtin_amdgcn_exp2f(x)
#else
#define EXP2F(x) __expf((x) * 0.6931471805599453f)
#endif

// tanh(z) given z2l = z * 2*log2(e): v_exp_f32 + v_add + v_rcp_f32 + v_fma.
// Exact limits (exp2->inf => 1, exp2->0 => -1); branchless.
__device__ __forceinline__ float tanh_scaled(float z2l) {
    const float t = EXP2F(z2l);
    const float r = __builtin_amdgcn_rcpf(t + 1.0f);
    return fmaf(-2.0f, r, 1.0f);
}
__device__ __forceinline__ float fast_tanh(float x) {
    return tanh_scaled(x * kTwoLog2E);
}

// Deep BSDE step loop. One wave owns 16 paths for all 32 steps.
// Layer 2 runs as h2^T = W2^T * h1^T on v_wmma_f32_16x16x32_f16 so that the
// layer-3 contraction over hidden units stays inside a lane.
__global__ __launch_bounds__(256) void bsde_wmma_kernel(
    const float* __restrict__ x0, const float* __restrict__ dW,
    const float* __restrict__ W1, const float* __restrict__ b1,
    const float* __restrict__ W2, const float* __restrict__ b2,
    const float* __restrict__ W3, const float* __restrict__ b3,
    float* __restrict__ partial)
{
    const int lane = threadIdx.x & 31;
    const int wave = threadIdx.x >> 5;
    const int half = lane >> 4;        // lane half selects K/M sub-ranges
    const int lp   = lane & 15;        // path-in-tile == WMMA N column
    const int tile = blockIdx.x * kWavesPerBlock + wave;
    const int path = tile * kPathsPerWave + lp;   // replicated across halves

    // ---- stage this wave's dW block (2KB, contiguous) into LDS ----
    // Global: row-major [path][step]; LDS: step-major [step][path] so the
    // per-step ds_load is conflict-free (lanes 0-15) / broadcast (16-31).
    __shared__ float sdW[kWavesPerBlock][kSteps * kPathsPerWave];
    {
        const float4* gsrc =
            (const float4*)(dW + (size_t)tile * (kPathsPerWave * kSteps));
        #pragma unroll
        for (int q = 0; q < 4; ++q) {                 // 4 x 32 lanes x 16B = 2KB
            const int    g4 = q * 32 + lane;          // float4 index
            const float4 v  = gsrc[g4];
            const int    p  = g4 >> 3;                // path-in-tile
            const int    i0 = (g4 & 7) * 4;           // first step of this quad
            sdW[wave][(i0 + 0) * 16 + p] = v.x;
            sdW[wave][(i0 + 1) * 16 + p] = v.y;
            sdW[wave][(i0 + 2) * 16 + p] = v.z;
            sdW[wave][(i0 + 3) * 16 + p] = v.w;
        }
    }

    // ---- A tiles: W2^T, 4 M-tiles (hidden-out) x 2 K-tiles (hidden-in) ----
    // A element e of lane: row m = 16j + lp, k = (e&7)|((e&8)<<1)|(half<<3)
    v16h A[4][2];
    #pragma unroll
    for (int j = 0; j < 4; ++j)
        #pragma unroll
        for (int t = 0; t < 2; ++t)
            #pragma unroll
            for (int e = 0; e < 16; ++e) {
                const int k   = (e & 7) | ((e & 8) << 1) | (half << 3);
                const int kin = t * 32 + k;           // hidden-in index
                const int mo  = j * 16 + lp;          // hidden-out index
                A[j][t][e] = (_Float16)W2[kin * kHid + mo];
            }

    // ---- per-lane layer-1 weights, pre-scaled by 2*log2(e) so layer 1 is
    //      fma -> v_exp_f32 directly ----
    _Float16 w1h[32], b1h[32];
    #pragma unroll
    for (int t = 0; t < 2; ++t)
        #pragma unroll
        for (int e = 0; e < 16; ++e) {
            const int kin = t * 32 + ((e & 7) | ((e & 8) << 1) | (half << 3));
            w1h[t * 16 + e] = (_Float16)(W1[kHid + kin] * kTwoLog2E);
            b1h[t * 16 + e] = (_Float16)(b1[kin] * kTwoLog2E);
        }

    // ---- b2 pre-imaged into WMMA C layout; W3 column 0 per lane (f16) ----
    v8f      cb2[4];
    _Float16 w3l[4][8];
    #pragma unroll
    for (int j = 0; j < 4; ++j)
        #pragma unroll
        for (int v = 0; v < 8; ++v) {
            const int m = j * 16 + v + half * 8;      // C/D row for VGPR v
            cb2[j][v] = b2[m];
            w3l[j][v] = (_Float16)W3[m * 2 + 0];      // y-head only; z is dead
        }
    const float b3y = b3[0];

    // ---- path state (replicated across lane halves) ----
    float x    = x0[path];
    float ylin = 1.0f;
    float lacc = 0.0f;

    __syncthreads();                                  // dW staging visible

    for (int i = 0; i < kSteps; ++i) {
        // Layer 1: h1 = tanh(x*W1[1,:] + b1), packed as B = h1^T (K x N).
        // Weights pre-scaled: exp2(fma(x,w1s,b1s)) == exp(2*(x*w1+b1)).
        v16h Bt[2];
        #pragma unroll
        for (int t = 0; t < 2; ++t)
            #pragma unroll
            for (int e = 0; e < 16; ++e) {
                const float h = tanh_scaled(fmaf(x, (float)w1h[t * 16 + e],
                                                    (float)b1h[t * 16 + e]));
                Bt[t][e] = (_Float16)h;
            }

        // Layer 2: h2pre^T = W2^T * h1^T + b2 (b2 enters as the C operand)
        // Layer 3: y = sum_m tanh(h2pre[m]) * W3[m,0]  (in-lane FMAs)
        float ysum = 0.0f;
        #pragma unroll
        for (int j = 0; j < 4; ++j) {
            v8f acc = __builtin_amdgcn_wmma_f32_16x16x32_f16(
                false, A[j][0], false, Bt[0], (short)0, cb2[j], false, false);
            acc = __builtin_amdgcn_wmma_f32_16x16x32_f16(
                false, A[j][1], false, Bt[1], (short)0, acc, false, false);
            #pragma unroll
            for (int v = 0; v < 8; ++v)
                ysum = fmaf(tanh_scaled(acc[v] * kTwoLog2E),
                            (float)w3l[j][v], ysum);
        }
        ysum += __shfl_xor(ysum, 16, 32);   // combine the two lane-half partials
        const float y   = ysum + b3y;
        const float ycv = y - ylin;
        lacc = fmaf(ycv, ycv, lacc);

        // tamed-Euler state update; dW from LDS (conflict-free/broadcast)
        const float dw   = sdW[wave][i * 16 + lp];
        float       mu   = kMuC * x;
        mu               = mu * __builtin_amdgcn_rcpf(fmaf(kDt, fabsf(mu), 1.0f));
        const float xold = x;
        x    = fmaf(mu, kDt, fmaf(kSigC * xold, dw, xold));
        ylin = fmaf(-fast_tanh(xold), kDt, ylin);
    }

    // loss = mean(ycv^2 over N*B) + 0.01*mean(ylin_T^2 over B);
    // each path lives in two lanes -> 0.5x
    const float invB = 1.0f / (float)kPaths;
    float contrib = 0.5f * (lacc * (invB / (float)kSteps)
                            + 0.01f * ylin * ylin * invB);

    #pragma unroll
    for (int s = 16; s >= 1; s >>= 1) contrib += __shfl_xor(contrib, s, 32);

    __shared__ float wsum[kWavesPerBlock];
    if (lane == 0) wsum[wave] = contrib;
    __syncthreads();
    if (threadIdx.x == 0) {
        float s = 0.0f;
        #pragma unroll
        for (int w = 0; w < kWavesPerBlock; ++w) s += wsum[w];
        partial[blockIdx.x] = s;
    }
}

// Deterministic fixed-order final reduction of the 512 block partials.
__global__ __launch_bounds__(256) void bsde_reduce_kernel(
    const float* __restrict__ partial, float* __restrict__ out, int n)
{
    __shared__ float sm[256];
    float s = 0.0f;
    for (int i = threadIdx.x; i < n; i += 256) s += partial[i];
    sm[threadIdx.x] = s;
    __syncthreads();
    for (int stride = 128; stride >= 1; stride >>= 1) {
        if ((int)threadIdx.x < stride) sm[threadIdx.x] += sm[threadIdx.x + stride];
        __syncthreads();
    }
    if (threadIdx.x == 0) out[0] = sm[0];
}

extern "C" void kernel_launch(void* const* d_in, const int* in_sizes, int n_in,
                              void* d_out, int out_size, void* d_ws, size_t ws_size,
                              hipStream_t stream) {
    (void)in_sizes; (void)n_in; (void)out_size; (void)ws_size;
    const float* x0 = (const float*)d_in[0];
    const float* dW = (const float*)d_in[1];
    const float* W1 = (const float*)d_in[2];
    const float* b1 = (const float*)d_in[3];
    const float* W2 = (const float*)d_in[4];
    const float* b2 = (const float*)d_in[5];
    const float* W3 = (const float*)d_in[6];
    const float* b3 = (const float*)d_in[7];
    float* partial = (float*)d_ws;          // 512 floats of scratch
    float* out     = (float*)d_out;

    bsde_wmma_kernel<<<kBlocks, 256, 0, stream>>>(x0, dW, W1, b1, W2, b2, W3, b3,
                                                  partial);
    bsde_reduce_kernel<<<1, 256, 0, stream>>>(partial, out, kBlocks);
}
